// PerspTransDetector_56143812493677
// MI455X (gfx1250) — matrix operand order repaired and checked by reading, not compiled, not run on hardware
//
#include <hip/hip_runtime.h>
#include <hip/hip_bf16.h>
#include <stdint.h>

#define NCAM 7
#define CCH  512
#define HF   135
#define WF   240
#define IMHW (HF*WF)      // 32400
#define GH   120
#define GW   360
#define NP   (GH*GW)      // 43200

typedef __bf16        v16bf __attribute__((ext_vector_type(16)));
typedef float         v8f   __attribute__((ext_vector_type(8)));
typedef unsigned int  v8u   __attribute__((ext_vector_type(8)));

static __device__ __forceinline__ unsigned short f2bf(float f) {
  unsigned int u = __float_as_uint(f);
  unsigned int r = u + 0x7FFFu + ((u >> 16) & 1u);   // round-to-nearest-even
  return (unsigned short)(r >> 16);
}
static __device__ __forceinline__ float bf2f(unsigned short h) {
  return __uint_as_float(((unsigned int)h) << 16);
}
static __device__ __forceinline__ v16bf asbf(v8u u) {
  return __builtin_bit_cast(v16bf, u);
}
// cross-half reduction: xor lane 16 via ds_swizzle (group-of-32, xor_mask=0x10)
static __device__ __forceinline__ float xor16(float v) {
  return __int_as_float(__builtin_amdgcn_ds_swizzle(__float_as_int(v), 0x401f));
}

// ---------------------------------------------------------------------------
// K0: transpose conv weights [oc][ic][3][3] fp32 -> [tap][oc][ic] bf16
// ---------------------------------------------------------------------------
__global__ void k_wt(const float* __restrict__ w, unsigned short* __restrict__ o) {
  int q = blockIdx.x * 256 + threadIdx.x;
  if (q >= 9 * 512 * 512) return;
  int pos = q / (512 * 512);
  int rem = q - pos * 512 * 512;
  int oc = rem >> 9;
  int ic = rem & 511;
  o[q] = f2bf(w[((size_t)oc * 512 + ic) * 9 + pos]);
}

// ---------------------------------------------------------------------------
// K1a: projective transform of world grid -> per-cam gather index (-1 = masked)
// ---------------------------------------------------------------------------
__global__ void k_proj(const float* __restrict__ proj, int* __restrict__ idx) {
  int p = blockIdx.x * 256 + threadIdx.x;
  if (p >= NP) return;
  float gh = (float)(p / GW);
  float gw = (float)(p % GW);
  for (int n = 0; n < NCAM; ++n) {
    const float* m = proj + n * 9;
    float cx = m[0] * gh + m[1] * gw + m[2];
    float cy = m[3] * gh + m[4] * gw + m[5];
    float cz = m[6] * gh + m[7] * gw + m[8];
    float x = cx / cz, y = cy / cz;
    int ix = (int)x;   // trunc toward zero == jnp.trunc + int cast
    int iy = (int)y;
    int s = -1;
    if (ix >= 0 && ix < WF && iy >= 0 && iy < HF) s = iy * WF + ix;
    idx[n * NP + p] = s;
  }
}

// ---------------------------------------------------------------------------
// K1b: masked gather + camera sum -> world map bf16 [C][P]
// ---------------------------------------------------------------------------
__global__ void k_world(const float* __restrict__ img, const int* __restrict__ idx,
                        unsigned short* __restrict__ world) {
  int p = blockIdx.x * 256 + threadIdx.x;
  int c = blockIdx.y;
  if (p >= NP) return;
  float acc = 0.f;
#pragma unroll
  for (int n = 0; n < NCAM; ++n) {
    int i = idx[n * NP + p];
    if (i >= 0) acc += img[(n * CCH + c) * IMHW + i];
  }
  world[(size_t)c * NP + p] = f2bf(acc * (1.0f / 7.0f));
}

// ---------------------------------------------------------------------------
// K2: fused image classifier: relu(W1[64x512] @ X + b1) -> W2[2x64] @ h
//     one wave = 64 oc x 16 px tile, 4 WMMA accumulators, K-loop 16x32
// ---------------------------------------------------------------------------
#define NTILES (NCAM * IMHW / 16)   // 14175 (exact)
#define APAD 520                     // 1040 B rows: 16B-aligned frags, bank spread
__launch_bounds__(256)
__global__ void k_imgcls(const float* __restrict__ img, const float* __restrict__ w1,
                         const float* __restrict__ b1, const float* __restrict__ w2,
                         float* __restrict__ out) {
  __shared__ unsigned short Alds[64 * APAD];
  __shared__ float w2s[128];
  __shared__ float b1s[64];
  int tid = threadIdx.x;
  for (int e = tid; e < 64 * 512; e += 256) {
    int m = e >> 9, k = e & 511;
    Alds[m * APAD + k] = f2bf(w1[e]);
  }
  if (tid < 128) w2s[tid] = w2[tid];
  if (tid < 64)  b1s[tid] = b1[tid];
  __syncthreads();

  int lane = tid & 31, wave = tid >> 5;
  int tile = blockIdx.x * 8 + wave;
  if (tile >= NTILES) return;               // wave-uniform exit
  int ln = lane & 15, hi = lane >> 4;

  // A fragments (weights), fixed for the whole kernel
  v8u afr[4];
#pragma unroll
  for (int t = 0; t < 4; ++t) {
    int m = 16 * t + ln;
#pragma unroll
    for (int j = 0; j < 8; ++j) {
      int k = 2 * j + (hi ? 16 : 0);
      afr[t][j] = *(const unsigned int*)&Alds[m * APAD + k];
    }
  }

  int pixc = tile * 16 + ln;                // tiles never cross camera bounds
  int n  = pixc / IMHW;
  int hw = pixc - n * IMHW;
  const float* bcol = img + (size_t)(n * CCH) * IMHW + hw;

  v8f acc[4];
#pragma unroll
  for (int t = 0; t < 4; ++t)
#pragma unroll
    for (int r = 0; r < 8; ++r) acc[t][r] = 0.f;

  for (int step = 0; step < 16; ++step) {
    v8u bu;
#pragma unroll
    for (int j = 0; j < 8; ++j) {
      int k = 32 * step + 2 * j + (hi ? 16 : 0);
      float f0 = bcol[(size_t)k * IMHW];
      float f1 = bcol[(size_t)(k + 1) * IMHW];
      bu[j] = (unsigned int)f2bf(f0) | ((unsigned int)f2bf(f1) << 16);
    }
    v16bf b = asbf(bu);
#pragma unroll
    for (int t = 0; t < 4; ++t)
      acc[t] = __builtin_amdgcn_wmma_f32_16x16x32_bf16(
          false, asbf(afr[t]), false, b, (short)0, acc[t], false, false);
  }

  // epilogue: bias + relu + 2-channel 1x1 conv, reduce across lane halves
  float p0 = 0.f, p1 = 0.f;
#pragma unroll
  for (int t = 0; t < 4; ++t)
#pragma unroll
    for (int r = 0; r < 8; ++r) {
      int m = 16 * t + r + (hi ? 8 : 0);    // D layout: M = r + 8*half
      float h = fmaxf(acc[t][r] + b1s[m], 0.f);
      p0 += w2s[m] * h;
      p1 += w2s[64 + m] * h;
    }
  p0 += xor16(p0);
  p1 += xor16(p1);
  if (hi == 0) {
    out[(n * 2 + 0) * IMHW + hw] = p0;
    out[(n * 2 + 1) * IMHW + hw] = p1;
  }
}

// ---------------------------------------------------------------------------
// K3: implicit-GEMM dilated 3x3 conv 512->512, bias+relu, bf16 in/out
//     block: 256 oc x 64 px; wave: 64 oc x 32 px (8 WMMA accumulators)
//     LDS rows padded to 40 halves (80 B): frags = 2 x ds_load_b128 each
// ---------------------------------------------------------------------------
#define KPAD 40
__launch_bounds__(256)
__global__ void k_conv(const unsigned short* __restrict__ in,
                       const unsigned short* __restrict__ Wt,   // [tap][oc][ic] bf16
                       const float* __restrict__ bias,
                       unsigned short* __restrict__ outbf, int dil) {
  __shared__ unsigned short Al[256 * KPAD];  // [oc_local][k]
  __shared__ unsigned short Bl[64 * KPAD];   // [px_local][k]
  int tid = threadIdx.x, lane = tid & 31, wave = tid >> 5;
  int wx = wave & 1;                // px group: 2 x 32
  int wy = wave >> 1;               // oc group: 4 x 64
  int ocb = blockIdx.y * 256;
  int pb  = blockIdx.x * 64;
  int ln = lane & 15, hi = lane >> 4;

  int skk  = tid & 31;              // B-stage: this thread's ic-in-chunk
  int spx0 = (tid >> 5) * 8;        // B-stage: 8 consecutive pixels

  v8f acc[4][2];
#pragma unroll
  for (int t = 0; t < 4; ++t)
#pragma unroll
    for (int u = 0; u < 2; ++u)
#pragma unroll
      for (int r = 0; r < 8; ++r) acc[t][u][r] = 0.f;

  for (int pos = 0; pos < 9; ++pos) {
    int dy = (pos / 3 - 1) * dil;
    int dx = (pos % 3 - 1) * dil;
    for (int icc = 0; icc < 512; icc += 32) {
      __syncthreads();
      { // stage A: 256 rows x 32 k; one contiguous 64B row per thread
        const uint4* s =
            (const uint4*)(Wt + ((size_t)pos * 512 + ocb + tid) * 512 + icc);
        uint4* d = (uint4*)&Al[tid * KPAD];
        d[0] = s[0]; d[1] = s[1]; d[2] = s[2]; d[3] = s[3];
      }
      { // stage B: 32 ic x 64 px gather with spatial shift + zero pad
#pragma unroll
        for (int i = 0; i < 8; ++i) {
          int pxl = spx0 + i;
          int p = pb + pxl;
          int gh = p / GW + dy;
          int gw = p % GW + dx;
          unsigned short v = 0;
          if ((unsigned)gh < (unsigned)GH && (unsigned)gw < (unsigned)GW)
            v = in[(size_t)(icc + skk) * NP + gh * GW + gw];
          Bl[pxl * KPAD + skk] = v;
        }
      }
      __syncthreads();

      v8u bu[2];
#pragma unroll
      for (int u = 0; u < 2; ++u)
#pragma unroll
        for (int j = 0; j < 8; ++j) {
          int k = 2 * j + (hi ? 16 : 0);
          bu[u][j] = *(const unsigned int*)&Bl[(wx * 32 + 16 * u + ln) * KPAD + k];
        }
      v16bf b0 = asbf(bu[0]);
      v16bf b1v = asbf(bu[1]);
#pragma unroll
      for (int t = 0; t < 4; ++t) {
        v8u au;
#pragma unroll
        for (int j = 0; j < 8; ++j) {
          int k = 2 * j + (hi ? 16 : 0);
          au[j] = *(const unsigned int*)&Al[(wy * 64 + 16 * t + ln) * KPAD + k];
        }
        v16bf a = asbf(au);
        acc[t][0] = __builtin_amdgcn_wmma_f32_16x16x32_bf16(
            false, a, false, b0, (short)0, acc[t][0], false, false);
        acc[t][1] = __builtin_amdgcn_wmma_f32_16x16x32_bf16(
            false, a, false, b1v, (short)0, acc[t][1], false, false);
      }
    }
  }

#pragma unroll
  for (int t = 0; t < 4; ++t)
#pragma unroll
    for (int u = 0; u < 2; ++u) {
      int pxg = pb + wx * 32 + 16 * u + ln;
#pragma unroll
      for (int r = 0; r < 8; ++r) {
        int oc = ocb + wy * 64 + 16 * t + r + (hi ? 8 : 0);
        float v = fmaxf(acc[t][u][r] + bias[oc], 0.f);
        outbf[(size_t)oc * NP + pxg] = f2bf(v);
      }
    }
}

// ---------------------------------------------------------------------------
// K4: final conv 512->1, dilation 4, no bias/relu, fp32 out (L2-resident input)
// ---------------------------------------------------------------------------
__global__ void k_conv3(const unsigned short* __restrict__ in,
                        const float* __restrict__ w3, float* __restrict__ out) {
  int p = blockIdx.x * 256 + threadIdx.x;
  if (p >= NP) return;
  int gh = p / GW, gw = p - (p / GW) * GW;
  float acc = 0.f;
  for (int pos = 0; pos < 9; ++pos) {
    int y = gh + (pos / 3 - 1) * 4;
    int x = gw + (pos % 3 - 1) * 4;
    bool ok = ((unsigned)y < (unsigned)GH) && ((unsigned)x < (unsigned)GW);
    int off = ok ? (y * GW + x) : 0;
    float t = 0.f;
    for (int ic = 0; ic < 512; ++ic)
      t += w3[ic * 9 + pos] * bf2f(in[(size_t)ic * NP + off]);
    acc += ok ? t : 0.f;
  }
  out[p] = acc;
}

// ---------------------------------------------------------------------------
extern "C" void kernel_launch(void* const* d_in, const int* in_sizes, int n_in,
                              void* d_out, int out_size, void* d_ws, size_t ws_size,
                              hipStream_t stream) {
  (void)in_sizes; (void)n_in; (void)out_size; (void)ws_size;
  const float* img   = (const float*)d_in[0];
  const float* proj  = (const float*)d_in[1];
  const float* ic_w1 = (const float*)d_in[2];
  const float* ic_b1 = (const float*)d_in[3];
  const float* ic_w2 = (const float*)d_in[4];
  const float* mc_w1 = (const float*)d_in[5];
  const float* mc_b1 = (const float*)d_in[6];
  const float* mc_w2 = (const float*)d_in[7];
  const float* mc_b2 = (const float*)d_in[8];
  const float* mc_w3 = (const float*)d_in[9];
  float* out_map = (float*)d_out;        // [43200]
  float* out_img = (float*)d_out + NP;   // [7*2*135*240]

  char* ws = (char*)d_ws;
  size_t off = 0;
  auto take = [&](size_t bytes) {
    char* p = ws + off;
    off = (off + bytes + 255) & ~(size_t)255;
    return p;
  };
  unsigned short* Wt1   = (unsigned short*)take((size_t)9 * 512 * 512 * 2);
  unsigned short* Wt2   = (unsigned short*)take((size_t)9 * 512 * 512 * 2);
  int*            idx   = (int*)take((size_t)NCAM * NP * 4);
  unsigned short* world = (unsigned short*)take((size_t)CCH * NP * 2);  // reused as h2
  unsigned short* h1    = (unsigned short*)take((size_t)CCH * NP * 2);

  k_wt<<<(9 * 512 * 512 + 255) / 256, 256, 0, stream>>>(mc_w1, Wt1);
  k_wt<<<(9 * 512 * 512 + 255) / 256, 256, 0, stream>>>(mc_w2, Wt2);
  k_proj<<<(NP + 255) / 256, 256, 0, stream>>>(proj, idx);
  k_world<<<dim3((NP + 255) / 256, CCH), 256, 0, stream>>>(img, idx, world);
  k_imgcls<<<(NTILES + 7) / 8, 256, 0, stream>>>(img, ic_w1, ic_b1, ic_w2, out_img);
  k_conv<<<dim3(NP / 64, 2), 256, 0, stream>>>(world, Wt1, mc_b1, h1, 1);
  k_conv<<<dim3(NP / 64, 2), 256, 0, stream>>>(h1, Wt2, mc_b2, world, 2);  // world := h2
  k_conv3<<<(NP + 255) / 256, 256, 0, stream>>>(world, mc_w3, out_map);
}
